// CAM_Module_55559696941194
// MI455X (gfx1250) — matrix operand order repaired and checked by reading, not compiled, not run on hardware
//
#include <hip/hip_runtime.h>
#include <hip/hip_bf16.h>

// CDNA5 / gfx1250: wave32, WMMA 16x16x32 bf16 -> f32 accumulate.

typedef __attribute__((ext_vector_type(16))) __bf16          v16bf;
typedef __attribute__((ext_vector_type(8)))  float           v8f;
typedef __attribute__((ext_vector_type(8)))  unsigned short  us8;
typedef __attribute__((ext_vector_type(16))) unsigned short  us16;

#define C_DIM  128
#define D_DIM  32
#define S_DIM  4096
#define KC     64     // k-chunk (phase 1) / s-chunk (phase 3)
#define P1_STRIDE 72  // LDS row stride (ushorts) phase-1 staging: 144B = 9*16B, conflict-free
#define P3_STRIDE 136 // LDS row stride (ushorts) phase-3 transposed staging: 272B
#define A_STRIDE  136 // LDS row stride (ushorts) attention matrix: 272B

__device__ __forceinline__ unsigned short f2bf(float f) {
  // round-to-nearest-even f32 -> bf16 (cold path: softmax store only)
  unsigned int u = __float_as_uint(f);
  u += 0x7fffu + ((u >> 16) & 1u);
  return (unsigned short)(u >> 16);
}
// Hot-path pack: two f32 -> packed bf16 pair in 3 VALU ops
// (2x v_add_u32 round-half-up, then one v_perm_b32 grabbing both high halves).
__device__ __forceinline__ unsigned int pack2(float a, float b) {
  const unsigned int ua = __float_as_uint(a) + 0x8000u;
  const unsigned int ub = __float_as_uint(b) + 0x8000u;
  // dst bytes (LSB..MSB) = ua[2], ua[3], ub[2], ub[3]  (src1=ua: sel 2,3; src0=ub: sel 6,7)
  return __builtin_amdgcn_perm(ub, ua, 0x07060302u);
}
// Build a 16-element bf16 fragment from two 8-element LDS chunks.
__device__ __forceinline__ v16bf frag16(const unsigned short* p0, const unsigned short* p1) {
  us8 lo = *(const us8*)p0;
  us8 hi = *(const us8*)p1;
  us16 t = __builtin_shufflevector(lo, hi, 0,1,2,3,4,5,6,7,8,9,10,11,12,13,14,15);
  return __builtin_bit_cast(v16bf, t);
}

// Pin a batch of fragments in VGPRs: forces all DS loads to issue (and their
// single counter drain to happen) BEFORE the WMMA chain, so the matrix ops run
// back-to-back instead of load/wait/wmma serialization.
#define PIN9(a0,b0,b1,b2,b3,b4,b5,b6,b7)                                     \
  asm volatile("" : "+v"(a0), "+v"(b0), "+v"(b1), "+v"(b2), "+v"(b3),        \
                    "+v"(b4), "+v"(b5), "+v"(b6), "+v"(b7))
#define PIN5(a0,b0,b1,b2,b3)                                                 \
  asm volatile("" : "+v"(a0), "+v"(b0), "+v"(b1), "+v"(b2), "+v"(b3))

__global__ __launch_bounds__(256)
void cam_fused_kernel(const float* __restrict__ x, const float* __restrict__ gamma,
                      float* __restrict__ out) {
  // 52 KB static LDS total
  __shared__ __align__(16) unsigned short stage[C_DIM * P1_STRIDE]; // 18,432 B (reused in phase 3)
  __shared__ __align__(16) unsigned short abf[C_DIM * A_STRIDE];    // 34,816 B attention (bf16)

  const int tid = threadIdx.x;
  const int w   = tid >> 5;        // wave id 0..7
  const int ln  = tid & 15;        // lane-in-half
  const int hf  = (tid >> 4) & 1;  // half-wave id

  const int n = blockIdx.x;        // slice = b*D + d
  const int b = n / D_DIM;
  const int d = n % D_DIM;
  const size_t rowStride = (size_t)D_DIM * S_DIM;                    // c-stride in x
  const float* __restrict__ xs = x + ((size_t)b * C_DIM * D_DIM + d) * (size_t)S_DIM;
  float* __restrict__ os = out + (size_t)n * C_DIM * (size_t)S_DIM;  // (B,D,C,S) contiguous

  const float g = gamma[0];
  const v8f vzero = {0.f,0.f,0.f,0.f,0.f,0.f,0.f,0.f};

  float4 vbuf[8];  // register double-buffer for global->LDS pipeline

  // ---------------- Phase 1: E = V * V^T (bf16 WMMA, f32 accumulate) ----------------
  // wave w owns rows [16w,16w+16) of E; 8 column tiles -> 8 accumulators.
  v8f acc[8];
#pragma unroll
  for (int i = 0; i < 8; ++i) acc[i] = vzero;

  // preload chunk 0 into registers
#pragma unroll
  for (int j = 0; j < 8; ++j) {
    const int i = tid + j * 256;
    vbuf[j] = *(const float4*)(xs + (size_t)(i >> 4) * rowStride + ((i & 15) << 2));
  }

  for (int k0 = 0; k0 < S_DIM; k0 += KC) {
    // convert BEFORE the barrier (pure VALU on registers, overlaps barrier wait)
    uint2 pk[8];
#pragma unroll
    for (int j = 0; j < 8; ++j) {
      pk[j].x = pack2(vbuf[j].x, vbuf[j].y);
      pk[j].y = pack2(vbuf[j].z, vbuf[j].w);
    }
    __syncthreads();  // stage free (previous-iteration readers done)
#pragma unroll
    for (int j = 0; j < 8; ++j) {
      const int i   = tid + j * 256;
      *(uint2*)&stage[(i >> 4) * P1_STRIDE + ((i & 15) << 2)] = pk[j];
    }
    __syncthreads();

    // issue next-chunk global loads BEFORE compute: latency hides behind WMMAs
    if (k0 + KC < S_DIM) {
#pragma unroll
      for (int j = 0; j < 8; ++j) {
        const int i   = tid + j * 256;
        const int row = i >> 4;
        const int kq  = (i & 15) << 2;
        vbuf[j] = *(const float4*)(xs + (size_t)row * rowStride + (k0 + KC + kq));
        if (k0 + 2 * KC < S_DIM)
          __builtin_prefetch(xs + (size_t)row * rowStride + (k0 + 2 * KC + kq), 0, 0);
      }
    }

    const int row_a = (w << 4) + ln;
#pragma unroll
    for (int kk = 0; kk < KC; kk += 32) {
      // A fragment: 16x32 bf16, lane half selects K sub-blocks
      const unsigned short* ab = &stage[row_a * P1_STRIDE + kk + hf * 8];
      v16bf afrag = frag16(ab, ab + 16);
      // batch all 8 B fragments
      v16bf bfr[8];
#pragma unroll
      for (int ct = 0; ct < 8; ++ct) {
        const unsigned short* bb = &stage[((ct << 4) + ln) * P1_STRIDE + kk + hf * 8];
        bfr[ct] = frag16(bb, bb + 16);
      }
      // pin the batch: one DS drain, then 8 WMMAs back-to-back
      PIN9(afrag, bfr[0], bfr[1], bfr[2], bfr[3], bfr[4], bfr[5], bfr[6], bfr[7]);
#pragma unroll
      for (int ct = 0; ct < 8; ++ct)
        acc[ct] = __builtin_amdgcn_wmma_f32_16x16x32_bf16(
            false, afrag, false, bfr[ct], (short)0, acc[ct], false, false);
    }
  }

  // ---------------- Phase 2: softmax(rowmax - E) == exp(rowmin - E)/sum, in registers ----
  // acc[ct][r] = E[16w + r + 8*hf][16ct + ln]; reduce across the 16-lane half via shfl_xor.
#pragma unroll
  for (int r = 0; r < 8; ++r) {
    float mn = acc[0][r];
#pragma unroll
    for (int ct = 1; ct < 8; ++ct) mn = fminf(mn, acc[ct][r]);
#pragma unroll
    for (int m = 8; m >= 1; m >>= 1) mn = fminf(mn, __shfl_xor(mn, m, 32));
    float p[8]; float sum = 0.f;
#pragma unroll
    for (int ct = 0; ct < 8; ++ct) { p[ct] = __expf(mn - acc[ct][r]); sum += p[ct]; }
#pragma unroll
    for (int m = 8; m >= 1; m >>= 1) sum += __shfl_xor(sum, m, 32);
    const float inv = 1.0f / sum;
    const int row = (w << 4) + r + (hf << 3);
#pragma unroll
    for (int ct = 0; ct < 8; ++ct)
      abf[row * A_STRIDE + (ct << 4) + ln] = f2bf(p[ct] * inv);
  }

  // ---------------- Phase 3: out = gamma * (A @ V) + V ----------------
  // wave w: column tile ct3 = w%4 (16 s-values), M half mh = w/4 (4 row tiles).
  const int ct3 = w & 3;
  const int mh  = w >> 2;
  // staging mapping: thread owns row-pair (mp2, mp2+1) and 16 s-values
  const int mp2 = (tid >> 2) << 1;
  const int sq0 = (tid & 3) << 4;

  // preload chunk 0
#pragma unroll
  for (int q = 0; q < 4; ++q) {
    vbuf[q]     = *(const float4*)(xs + (size_t)mp2       * rowStride + (sq0 + (q << 2)));
    vbuf[4 + q] = *(const float4*)(xs + (size_t)(mp2 + 1) * rowStride + (sq0 + (q << 2)));
  }

  for (int s0 = 0; s0 < S_DIM; s0 += KC) {
    // convert BEFORE the barrier: pack row-pair (m, m+1) per s into b32 words
    unsigned int pk3[16];
#pragma unroll
    for (int q = 0; q < 4; ++q) {
      const float4 a  = vbuf[q];
      const float4 bq = vbuf[4 + q];
      pk3[4 * q + 0] = pack2(a.x, bq.x);
      pk3[4 * q + 1] = pack2(a.y, bq.y);
      pk3[4 * q + 2] = pack2(a.z, bq.z);
      pk3[4 * q + 3] = pack2(a.w, bq.w);
    }
    __syncthreads();  // all waves done reading stage
    // store transposed: stage[s_local][m]
#pragma unroll
    for (int e = 0; e < 16; ++e)
      *(unsigned int*)&stage[(sq0 + e) * P3_STRIDE + mp2] = pk3[e];
    __syncthreads();

    // issue next-chunk loads before compute
    if (s0 + KC < S_DIM) {
#pragma unroll
      for (int q = 0; q < 4; ++q) {
        vbuf[q]     = *(const float4*)(xs + (size_t)mp2       * rowStride + (s0 + KC + sq0 + (q << 2)));
        vbuf[4 + q] = *(const float4*)(xs + (size_t)(mp2 + 1) * rowStride + (s0 + KC + sq0 + (q << 2)));
        if (s0 + 2 * KC < S_DIM)
          __builtin_prefetch(xs + (size_t)mp2 * rowStride + (s0 + 2 * KC + sq0 + (q << 2)), 0, 0);
      }
    }

    v8f acc2[4];
#pragma unroll
    for (int t = 0; t < 4; ++t) acc2[t] = vzero;

#pragma unroll
    for (int kk = 0; kk < 128; kk += 32) {
      // B fragment: V[m, s] with K=m contiguous in transposed stage
      const unsigned short* bb = &stage[((ct3 << 4) + ln) * P3_STRIDE + kk + hf * 8];
      v16bf bfrag = frag16(bb, bb + 16);
      // batch the 4 A fragments
      v16bf afr[4];
#pragma unroll
      for (int t = 0; t < 4; ++t) {
        const int mt = (mh << 2) + t;
        const unsigned short* ab = &abf[((mt << 4) + ln) * A_STRIDE + kk + hf * 8];
        afr[t] = frag16(ab, ab + 16);
      }
      PIN5(bfrag, afr[0], afr[1], afr[2], afr[3]);
#pragma unroll
      for (int t = 0; t < 4; ++t)
        acc2[t] = __builtin_amdgcn_wmma_f32_16x16x32_bf16(
            false, afr[t], false, bfrag, (short)0, acc2[t], false, false);
    }

    // fused epilogue: gamma*acc + residual (residual re-read hits L2), f32 output
    const int sg = s0 + (ct3 << 4) + ln;
#pragma unroll
    for (int t = 0; t < 4; ++t) {
#pragma unroll
      for (int r = 0; r < 8; ++r) {
        const int c = (((mh << 2) + t) << 4) + r + (hf << 3);
        const float resid = xs[(size_t)c * rowStride + sg];
        os[(size_t)c * (size_t)S_DIM + sg] = g * acc2[t][r] + resid;
      }
    }
  }
}

extern "C" void kernel_launch(void* const* d_in, const int* in_sizes, int n_in,
                              void* d_out, int out_size, void* d_ws, size_t ws_size,
                              hipStream_t stream) {
  (void)in_sizes; (void)n_in; (void)d_ws; (void)ws_size; (void)out_size;
  const float* x     = (const float*)d_in[0];
  const float* gamma = (const float*)d_in[1];
  float* out         = (float*)d_out;
  // one workgroup per (b,d) slice: 4*32 = 128 blocks, 256 threads (8 waves)
  cam_fused_kernel<<<dim3(128), dim3(256), 0, stream>>>(x, gamma, out);
}
